// DLDMD_70428873719930
// MI455X (gfx1250) — matrix operand order repaired and checked by reading, not compiled, not run on hardware
//
#include <hip/hip_runtime.h>

// ---------------------------------------------------------------------------
// Types
// ---------------------------------------------------------------------------
typedef _Float16 h16;
typedef __attribute__((ext_vector_type(16))) _Float16 v16h;
typedef __attribute__((ext_vector_type(8)))  _Float16 v8h;
typedef __attribute__((ext_vector_type(8)))  float    v8f;

struct c32 { float re, im; };
__device__ inline c32 cadd(c32 a, c32 b){ return {a.re+b.re, a.im+b.im}; }
__device__ inline c32 csub(c32 a, c32 b){ return {a.re-b.re, a.im-b.im}; }
__device__ inline c32 cmul(c32 a, c32 b){ return {a.re*b.re - a.im*b.im, a.re*b.im + a.im*b.re}; }
__device__ inline c32 cdiv(c32 a, c32 b){
  float dn = b.re*b.re + b.im*b.im;
  if (dn < 1e-30f) { b.re += 1e-15f; dn = b.re*b.re + b.im*b.im; }
  float inv = 1.0f / dn;
  return {(a.re*b.re + a.im*b.im)*inv, (a.im*b.re - a.re*b.im)*inv};
}

#define LDS_LD 264   // 256 + 8 halves pad (528B rows: 16B aligned, bank-spread)

// ---------------------------------------------------------------------------
// WMMA fragment helpers (CDNA5 16-bit layouts, wave32)
// A 16x32 f16: lane m=l&15, hi=l>>4; halves = [k0+8*hi .. +7] ++ [k0+16+8*hi .. +7]
// B 32x16 f16: lane n=l&15, hi=l>>4; halves = WT[n][k0+8*hi .. +7] ++ WT[n][k0+16+8*hi .. +7] (WT = W^T, N-major)
// ---------------------------------------------------------------------------
__device__ inline v16h load_a_frag(const h16* rowbase, int k0, int hi){
  const v8h* p0 = (const v8h*)(rowbase + k0 + 8*hi);
  const v8h* p1 = (const v8h*)(rowbase + k0 + 16 + 8*hi);
  v8h a0 = *p0, a1 = *p1;
  v16h r;
#pragma unroll
  for (int i = 0; i < 8; i++){ r[i] = a0[i]; r[i+8] = a1[i]; }
  return r;
}

__device__ inline v16h load_b_frag(const h16* WT, int ldk, int n, int k0, int hi){
  const v8h* p = (const v8h*)(WT + (size_t)n*ldk + k0 + 8*hi);
  v8h b0 = p[0], b1 = p[2];
  v16h r;
#pragma unroll
  for (int i = 0; i < 8; i++){ r[i] = b0[i]; r[i+8] = b1[i]; }
  return r;
}

__device__ inline v8f wmma_f16(v16h a, v16h b, v8f c){
  return __builtin_amdgcn_wmma_f32_16x16x32_f16(false, a, false, b, (short)0, c, false, false);
}

// D 16x16 f32: lane n=l&15; VGPR r holds row msub + r + 8*(l>>4)
__device__ inline void store_d_relu_f16(v8f acc, h16* buf, int msub, int hi,
                                        int n0, int l15, const float* bias){
  int n = n0 + l15;
  float bv = bias[n];
#pragma unroll
  for (int r = 0; r < 8; r++){
    int m = msub + r + 8*hi;
    buf[m*LDS_LD + n] = (h16)fmaxf(acc[r] + bv, 0.0f);
  }
}

// ---------------------------------------------------------------------------
// Weight transpose + f32->f16 convert:  W (K x N)  ->  WT (N x K) f16
// ---------------------------------------------------------------------------
__global__ void k_transcvt(const float* __restrict__ W, h16* __restrict__ WT,
                           int K, int N){
  int idx = blockIdx.x * blockDim.x + threadIdx.x;
  if (idx >= K * N) return;
  int k = idx / N, n = idx - k * N;
  WT[(size_t)n * K + k] = (h16)W[idx];
}

// ---------------------------------------------------------------------------
// Encoder MLP: rows x 2 -> 256 -> (256)x3 -> 64.  One block = 32 rows.
// ---------------------------------------------------------------------------
__global__ __launch_bounds__(256)
void k_encoder(const float* __restrict__ x, const float* __restrict__ Win,
               const float* __restrict__ bin, const h16* __restrict__ WhT,
               const float* __restrict__ bh, const h16* __restrict__ WoutT,
               const float* __restrict__ bout, float* __restrict__ y)
{
  __shared__ __align__(16) h16 bufA[32 * LDS_LD];
  __shared__ __align__(16) h16 bufB[32 * LDS_LD];
  const int tid  = threadIdx.x;
  const int lane = tid & 31, wv = tid >> 5;
  const int l15  = lane & 15, hi = lane >> 4;
  const int msub = (wv & 1) * 16;
  const int ngrp = wv >> 1;                 // 0..3 -> 64-col group
  const long row0 = (long)blockIdx.x * 32;

  // input layer (K=2): plain FMA
  for (int idx = tid; idx < 32 * 256; idx += 256){
    int m = idx >> 8, n = idx & 255;
    const float* xr = x + (row0 + m) * 2;
    float h = xr[0] * Win[n] + xr[1] * Win[256 + n] + bin[n];
    bufA[m * LDS_LD + n] = (h16)fmaxf(h, 0.0f);
  }
  __syncthreads();

  h16* cur = bufA; h16* nxt = bufB;
  for (int layer = 0; layer < 3; layer++){
    const h16* WT = WhT + layer * 65536;
    const float* bias = bh + layer * 256;
    v8f acc0 = {}, acc1 = {}, acc2 = {}, acc3 = {};
    const h16* abase = cur + (msub + l15) * LDS_LD;
    const int nb = ngrp * 64;
    for (int k0 = 0; k0 < 256; k0 += 32){
      __builtin_prefetch(WT + (size_t)(nb + l15) * 256 + k0 + 64, 0, 1);
      v16h a = load_a_frag(abase, k0, hi);
      acc0 = wmma_f16(a, load_b_frag(WT, 256, nb +  0 + l15, k0, hi), acc0);
      acc1 = wmma_f16(a, load_b_frag(WT, 256, nb + 16 + l15, k0, hi), acc1);
      acc2 = wmma_f16(a, load_b_frag(WT, 256, nb + 32 + l15, k0, hi), acc2);
      acc3 = wmma_f16(a, load_b_frag(WT, 256, nb + 48 + l15, k0, hi), acc3);
    }
    store_d_relu_f16(acc0, nxt, msub, hi, nb +  0, l15, bias);
    store_d_relu_f16(acc1, nxt, msub, hi, nb + 16, l15, bias);
    store_d_relu_f16(acc2, nxt, msub, hi, nb + 32, l15, bias);
    store_d_relu_f16(acc3, nxt, msub, hi, nb + 48, l15, bias);
    __syncthreads();
    h16* t = cur; cur = nxt; nxt = t;
  }

  // output layer 256 -> 64 (no relu), each wave does one 16-col tile
  {
    v8f acc = {};
    const h16* abase = cur + (msub + l15) * LDS_LD;
    int n0 = ngrp * 16;
    for (int k0 = 0; k0 < 256; k0 += 32){
      v16h a = load_a_frag(abase, k0, hi);
      acc = wmma_f16(a, load_b_frag(WoutT, 256, n0 + l15, k0, hi), acc);
    }
    int n = n0 + l15;
    float bv = bout[n];
#pragma unroll
    for (int r = 0; r < 8; r++){
      int m = msub + r + 8*hi;
      y[(row0 + m) * 64 + n] = acc[r] + bv;
    }
  }
}

// ---------------------------------------------------------------------------
// Decoder MLP: rows x 64 -> 256 -> (256)x3 -> 2.  One block = 32 rows.
// ---------------------------------------------------------------------------
__global__ __launch_bounds__(256)
void k_decoder(const float* __restrict__ in, const h16* __restrict__ WinT,
               const float* __restrict__ bin, const h16* __restrict__ WhT,
               const float* __restrict__ bh, const float* __restrict__ Wout,
               const float* __restrict__ bout, float* __restrict__ out)
{
  __shared__ __align__(16) h16 bufI[32 * 72];
  __shared__ __align__(16) h16 bufA[32 * LDS_LD];
  __shared__ __align__(16) h16 bufB[32 * LDS_LD];
  const int tid  = threadIdx.x;
  const int lane = tid & 31, wv = tid >> 5;
  const int l15  = lane & 15, hi = lane >> 4;
  const int msub = (wv & 1) * 16;
  const int ngrp = wv >> 1;
  const long row0 = (long)blockIdx.x * 32;

  for (int idx = tid; idx < 32 * 64; idx += 256){
    int m = idx >> 6, n = idx & 63;
    bufI[m * 72 + n] = (h16)in[(row0 + m) * 64 + n];
  }
  __syncthreads();

  // layer0: 64 -> 256 via WMMA (K loop of 2)
  {
    v8f acc0 = {}, acc1 = {}, acc2 = {}, acc3 = {};
    const h16* abase = bufI + (msub + l15) * 72;
    const int nb = ngrp * 64;
    for (int k0 = 0; k0 < 64; k0 += 32){
      v16h a = load_a_frag(abase, k0, hi);
      acc0 = wmma_f16(a, load_b_frag(WinT, 64, nb +  0 + l15, k0, hi), acc0);
      acc1 = wmma_f16(a, load_b_frag(WinT, 64, nb + 16 + l15, k0, hi), acc1);
      acc2 = wmma_f16(a, load_b_frag(WinT, 64, nb + 32 + l15, k0, hi), acc2);
      acc3 = wmma_f16(a, load_b_frag(WinT, 64, nb + 48 + l15, k0, hi), acc3);
    }
    store_d_relu_f16(acc0, bufA, msub, hi, nb +  0, l15, bin);
    store_d_relu_f16(acc1, bufA, msub, hi, nb + 16, l15, bin);
    store_d_relu_f16(acc2, bufA, msub, hi, nb + 32, l15, bin);
    store_d_relu_f16(acc3, bufA, msub, hi, nb + 48, l15, bin);
    __syncthreads();
  }

  h16* cur = bufA; h16* nxt = bufB;
  for (int layer = 0; layer < 3; layer++){
    const h16* WT = WhT + layer * 65536;
    const float* bias = bh + layer * 256;
    v8f acc0 = {}, acc1 = {}, acc2 = {}, acc3 = {};
    const h16* abase = cur + (msub + l15) * LDS_LD;
    const int nb = ngrp * 64;
    for (int k0 = 0; k0 < 256; k0 += 32){
      __builtin_prefetch(WT + (size_t)(nb + l15) * 256 + k0 + 64, 0, 1);
      v16h a = load_a_frag(abase, k0, hi);
      acc0 = wmma_f16(a, load_b_frag(WT, 256, nb +  0 + l15, k0, hi), acc0);
      acc1 = wmma_f16(a, load_b_frag(WT, 256, nb + 16 + l15, k0, hi), acc1);
      acc2 = wmma_f16(a, load_b_frag(WT, 256, nb + 32 + l15, k0, hi), acc2);
      acc3 = wmma_f16(a, load_b_frag(WT, 256, nb + 48 + l15, k0, hi), acc3);
    }
    store_d_relu_f16(acc0, nxt, msub, hi, nb +  0, l15, bias);
    store_d_relu_f16(acc1, nxt, msub, hi, nb + 16, l15, bias);
    store_d_relu_f16(acc2, nxt, msub, hi, nb + 32, l15, bias);
    store_d_relu_f16(acc3, nxt, msub, hi, nb + 48, l15, bias);
    __syncthreads();
    h16* t = cur; cur = nxt; nxt = t;
  }

  // output layer 256 -> 2, VALU dot products
  if (tid < 64){
    int m = tid >> 1, c = tid & 1;
    const h16* rp = cur + m * LDS_LD;
    float s = 0.0f;
    for (int k = 0; k < 256; k++) s += (float)rp[k] * Wout[k * 2 + c];
    out[(row0 + m) * 2 + c] = s + bout[c];
  }
}

// ---------------------------------------------------------------------------
// G = Ym Ym^T, C = Yp Ym^T   (per batch; Ym col t = y[b][t][:], t<511)
// ---------------------------------------------------------------------------
__global__ __launch_bounds__(256)
void k_gc(const float* __restrict__ y, float* __restrict__ G, float* __restrict__ C)
{
  const int b = blockIdx.x, tid = threadIdx.x;
  __shared__ float s[65][64];
  float g[16], c[16];
#pragma unroll
  for (int p = 0; p < 16; p++){ g[p] = 0.0f; c[p] = 0.0f; }
  const float* yb = y + (size_t)b * 512 * 64;
  const int pbase = tid * 16;
  for (int chunk = 0; chunk < 8; chunk++){
    int base = chunk * 64;
    for (int i = tid; i < 65 * 64; i += 256){
      int r = i >> 6, cc = i & 63, gr = base + r;
      if (gr < 512) s[r][cc] = yb[(size_t)gr * 64 + cc];
    }
    __syncthreads();
    for (int lt = 0; lt < 64; lt++){
      int gt = base + lt;
      if (gt >= 511) break;
#pragma unroll
      for (int p = 0; p < 16; p++){
        int i = (pbase + p) >> 6, j = (pbase + p) & 63;
        g[p] += s[lt][i] * s[lt][j];
        c[p] += s[lt + 1][i] * s[lt][j];
      }
    }
    __syncthreads();
  }
#pragma unroll
  for (int p = 0; p < 16; p++){
    G[(size_t)b * 4096 + pbase + p] = g[p];
    C[(size_t)b * 4096 + pbase + p] = c[p];
  }
}

// ---------------------------------------------------------------------------
// A = C * G^{-1}  (Gauss-Jordan on [G|I] in LDS, then VALU matmul)
// ---------------------------------------------------------------------------
__global__ __launch_bounds__(256)
void k_solveA(const float* __restrict__ G, const float* __restrict__ C,
              float* __restrict__ A)
{
  const int b = blockIdx.x, tid = threadIdx.x;
  __shared__ float aug[64][130];
  __shared__ float fac[64];
  __shared__ int spiv;
  for (int idx = tid; idx < 4096; idx += 256){
    int i = idx >> 6, j = idx & 63;
    aug[i][j] = G[(size_t)b * 4096 + idx];
    aug[i][64 + j] = (i == j) ? 1.0f : 0.0f;
  }
  __syncthreads();
  for (int k = 0; k < 64; k++){
    if (tid == 0){
      int best = k; float bm = -1.0f;
      for (int r = k; r < 64; r++){
        float mg = fabsf(aug[r][k]);
        if (mg > bm){ bm = mg; best = r; }
      }
      spiv = best;
    }
    __syncthreads();
    int p = spiv;
    if (p != k){
      for (int j = tid; j < 128; j += 256){
        float tv = aug[k][j]; aug[k][j] = aug[p][j]; aug[p][j] = tv;
      }
    }
    __syncthreads();
    float pv = aug[k][k];
    if (fabsf(pv) < 1e-30f) pv = (pv >= 0.0f) ? 1e-30f : -1e-30f;
    float inv = 1.0f / pv;
    for (int j = tid; j < 128; j += 256) aug[k][j] *= inv;
    __syncthreads();
    if (tid < 64) fac[tid] = (tid == k) ? 0.0f : aug[tid][k];
    __syncthreads();
    for (int idx = tid; idx < 64 * 128; idx += 256){
      int r = idx >> 7, j = idx & 127;
      if (r != k) aug[r][j] -= fac[r] * aug[k][j];
    }
    __syncthreads();
  }
  const float* Cb = C + (size_t)b * 4096;
  for (int idx = tid; idx < 4096; idx += 256){
    int i = idx >> 6, j = idx & 63;
    float s = 0.0f;
    for (int k2 = 0; k2 < 64; k2++) s += Cb[i * 64 + k2] * aug[k2][64 + j];
    A[(size_t)b * 4096 + idx] = s;
  }
}

// ---------------------------------------------------------------------------
// Real Schur (Hessenberg + Francis double-shift QR), one thread per batch.
// Writes quasi-triangular T, orthogonal Q, and complex eigenvalues.
// ---------------------------------------------------------------------------
__global__ void k_schur(const float* __restrict__ Ain, float* __restrict__ Tm,
                        float* __restrict__ Qm, float* __restrict__ evals)
{
  int b = blockIdx.x * blockDim.x + threadIdx.x;
  if (b >= 256) return;
  float* Hp = Tm + (size_t)b * 4096;
  float* Qp = Qm + (size_t)b * 4096;
  const float* Ap = Ain + (size_t)b * 4096;
#define HH(i,j) Hp[((i) << 6) + (j)]
#define QQ(i,j) Qp[((i) << 6) + (j)]
  for (int i = 0; i < 4096; i++) Hp[i] = Ap[i];
  for (int i = 0; i < 64; i++)
    for (int j = 0; j < 64; j++) QQ(i, j) = (i == j) ? 1.0f : 0.0f;

  float v[64];
  // Householder Hessenberg reduction
  for (int k = 0; k < 62; k++){
    float xn = 0.0f;
    for (int i = k + 1; i < 64; i++){ float t = HH(i, k); xn += t * t; }
    xn = sqrtf(xn);
    if (xn < 1e-30f) continue;
    float a0 = HH(k + 1, k);
    float alpha = (a0 >= 0.0f) ? -xn : xn;
    v[k + 1] = a0 - alpha;
    for (int i = k + 2; i < 64; i++) v[i] = HH(i, k);
    float vns = 0.0f;
    for (int i = k + 1; i < 64; i++) vns += v[i] * v[i];
    if (vns < 1e-30f) continue;
    float beta = 2.0f / vns;
    for (int j = k; j < 64; j++){
      float s = 0.0f;
      for (int i = k + 1; i < 64; i++) s += v[i] * HH(i, j);
      s *= beta;
      for (int i = k + 1; i < 64; i++) HH(i, j) -= v[i] * s;
    }
    for (int i = 0; i < 64; i++){
      float s = 0.0f;
      for (int j = k + 1; j < 64; j++) s += HH(i, j) * v[j];
      s *= beta;
      for (int j = k + 1; j < 64; j++) HH(i, j) -= s * v[j];
    }
    for (int i = 0; i < 64; i++){
      float s = 0.0f;
      for (int j = k + 1; j < 64; j++) s += QQ(i, j) * v[j];
      s *= beta;
      for (int j = k + 1; j < 64; j++) QQ(i, j) -= s * v[j];
    }
    for (int i = k + 2; i < 64; i++) HH(i, k) = 0.0f;
  }

  // Francis double-shift QR with bulge chasing
  int hi = 63, iter = 0;
  const float ulp = 1e-7f;
  while (hi > 0 && iter < 3000){
    int l = hi;
    while (l > 0){
      float s = fabsf(HH(l - 1, l - 1)) + fabsf(HH(l, l));
      if (s < 1e-30f) s = 1.0f;
      if (fabsf(HH(l, l - 1)) < ulp * s){ HH(l, l - 1) = 0.0f; break; }
      l--;
    }
    if (l == hi){ hi--; continue; }
    if (l == hi - 1){ hi -= 2; continue; }
    int lo = l;
    float sh = HH(hi - 1, hi - 1) + HH(hi, hi);
    float dt = HH(hi - 1, hi - 1) * HH(hi, hi) - HH(hi - 1, hi) * HH(hi, hi - 1);
    if ((iter & 31) == 30){
      float w1 = fabsf(HH(hi, hi - 1)) + fabsf(HH(hi - 1, hi - 2));
      sh = 1.5f * w1; dt = w1 * w1;
    }
    float xx = HH(lo, lo) * HH(lo, lo) + HH(lo, lo + 1) * HH(lo + 1, lo)
             - sh * HH(lo, lo) + dt;
    float yy = HH(lo + 1, lo) * (HH(lo, lo) + HH(lo + 1, lo + 1) - sh);
    float zz = HH(lo + 1, lo) * HH(lo + 2, lo + 1);
    for (int m = lo; m <= hi - 2; m++){
      float nr = sqrtf(xx * xx + yy * yy + zz * zz);
      if (nr > 1e-30f){
        float alpha = (xx >= 0.0f) ? -nr : nr;
        float v0 = xx - alpha, v1 = yy, v2 = zz;
        float vns = v0 * v0 + v1 * v1 + v2 * v2;
        if (vns > 1e-30f){
          float beta = 2.0f / vns;
          int js = (m > lo) ? (m - 1) : lo;
          for (int j = js; j < 64; j++){
            float sm = v0 * HH(m, j) + v1 * HH(m + 1, j) + v2 * HH(m + 2, j);
            sm *= beta;
            HH(m, j) -= v0 * sm; HH(m + 1, j) -= v1 * sm; HH(m + 2, j) -= v2 * sm;
          }
          int ie = (hi < m + 3) ? hi : (m + 3);
          for (int i = 0; i <= ie; i++){
            float sm = v0 * HH(i, m) + v1 * HH(i, m + 1) + v2 * HH(i, m + 2);
            sm *= beta;
            HH(i, m) -= sm * v0; HH(i, m + 1) -= sm * v1; HH(i, m + 2) -= sm * v2;
          }
          for (int i = 0; i < 64; i++){
            float sm = v0 * QQ(i, m) + v1 * QQ(i, m + 1) + v2 * QQ(i, m + 2);
            sm *= beta;
            QQ(i, m) -= sm * v0; QQ(i, m + 1) -= sm * v1; QQ(i, m + 2) -= sm * v2;
          }
        }
      }
      xx = HH(m + 1, m); yy = HH(m + 2, m);
      zz = (m < hi - 2) ? HH(m + 3, m) : 0.0f;
    }
    { // trailing 2x2 rotation
      float nr = sqrtf(xx * xx + yy * yy);
      if (nr > 1e-30f){
        float alpha = (xx >= 0.0f) ? -nr : nr;
        float v0 = xx - alpha, v1 = yy;
        float vns = v0 * v0 + v1 * v1;
        if (vns > 1e-30f){
          float beta = 2.0f / vns;
          int m = hi - 1;
          for (int j = m - 1; j < 64; j++){
            float sm = v0 * HH(m, j) + v1 * HH(m + 1, j); sm *= beta;
            HH(m, j) -= v0 * sm; HH(m + 1, j) -= v1 * sm;
          }
          for (int i = 0; i <= hi; i++){
            float sm = v0 * HH(i, m) + v1 * HH(i, m + 1); sm *= beta;
            HH(i, m) -= sm * v0; HH(i, m + 1) -= sm * v1;
          }
          for (int i = 0; i < 64; i++){
            float sm = v0 * QQ(i, m) + v1 * QQ(i, m + 1); sm *= beta;
            QQ(i, m) -= sm * v0; QQ(i, m + 1) -= sm * v1;
          }
        }
      }
    }
    iter++;
  }

  // eigenvalues from 1x1 / 2x2 diagonal blocks
  float* ev = evals + (size_t)b * 128;
  int k = 0;
  while (k < 64){
    bool split;
    if (k == 63) split = true;
    else {
      float s = fabsf(HH(k, k)) + fabsf(HH(k + 1, k + 1));
      if (s < 1e-30f) s = 1.0f;
      split = fabsf(HH(k + 1, k)) < 1e-6f * s;
    }
    if (split){ ev[2 * k] = HH(k, k); ev[2 * k + 1] = 0.0f; k++; }
    else {
      float tr = HH(k, k) + HH(k + 1, k + 1);
      float det = HH(k, k) * HH(k + 1, k + 1) - HH(k, k + 1) * HH(k + 1, k);
      float disc = 0.25f * tr * tr - det;
      if (disc < 0.0f){
        float im = sqrtf(-disc);
        ev[2 * k] = 0.5f * tr;     ev[2 * k + 1] = im;
        ev[2 * k + 2] = 0.5f * tr; ev[2 * k + 3] = -im;
      } else {
        float sq = sqrtf(disc);
        ev[2 * k] = 0.5f * tr + sq;     ev[2 * k + 1] = 0.0f;
        ev[2 * k + 2] = 0.5f * tr - sq; ev[2 * k + 3] = 0.0f;
      }
      k += 2;
    }
  }
#undef HH
#undef QQ
}

// ---------------------------------------------------------------------------
// Eigenvectors: complex back-substitution on T, rotate by Q. Thread per batch.
// evecs (B,64,64) complex interleaved: evecs[b][i][k] = component i of vec k.
// ---------------------------------------------------------------------------
__global__ void k_evec(const float* __restrict__ Tm, const float* __restrict__ Qm,
                       const float* __restrict__ evals, float* __restrict__ evecs)
{
  int b = blockIdx.x * blockDim.x + threadIdx.x;
  if (b >= 256) return;
  const float* T = Tm + (size_t)b * 4096;
  const float* Q = Qm + (size_t)b * 4096;
  c32 w[64], vv[64];
  for (int k = 0; k < 64; k++){
    c32 lam = {evals[b * 128 + 2 * k], evals[b * 128 + 2 * k + 1]};
    w[k] = {1.0f, 0.0f};
    for (int j = k - 1; j >= 0; j--){
      c32 s = {0.0f, 0.0f};
      for (int m = j + 1; m <= k; m++){
        float tv = T[j * 64 + m];
        s.re += tv * w[m].re; s.im += tv * w[m].im;
      }
      c32 den = {lam.re - T[j * 64 + j], lam.im};
      c32 ns = {-s.re, -s.im};
      w[j] = cdiv(ns, den);
    }
    float nrm = 0.0f;
    for (int i = 0; i < 64; i++){
      c32 s = {0.0f, 0.0f};
      for (int j = 0; j <= k; j++){
        float q = Q[i * 64 + j];
        s.re += q * w[j].re; s.im += q * w[j].im;
      }
      vv[i] = s;
      nrm += s.re * s.re + s.im * s.im;
    }
    float sc = rsqrtf(fmaxf(nrm, 1e-30f));
    for (int i = 0; i < 64; i++){
      size_t o = (((size_t)b * 64 + i) * 64 + k) * 2;
      evecs[o] = vv[i].re * sc;
      evecs[o + 1] = vv[i].im * sc;
    }
  }
}

// ---------------------------------------------------------------------------
// Phi = evecs^{-1} Ym : complex LU (partial pivot) in LDS, 511 RHS per batch.
// ---------------------------------------------------------------------------
__global__ __launch_bounds__(256)
void k_phi(const float* __restrict__ evecs, const float* __restrict__ y,
           float* __restrict__ Phi)
{
  const int b = blockIdx.x, tid = threadIdx.x;
  __shared__ c32 M[64][64];
  __shared__ int perm[64];
  __shared__ int spiv;
  const float* E = evecs + (size_t)b * 4096 * 2;
  for (int idx = tid; idx < 4096; idx += 256){
    int i = idx >> 6, j = idx & 63;
    M[i][j] = {E[idx * 2], E[idx * 2 + 1]};
  }
  if (tid < 64) perm[tid] = tid;
  __syncthreads();
  for (int k = 0; k < 64; k++){
    if (tid == 0){
      int best = k; float bm = -1.0f;
      for (int r = k; r < 64; r++){
        float mg = fabsf(M[r][k].re) + fabsf(M[r][k].im);
        if (mg > bm){ bm = mg; best = r; }
      }
      spiv = best;
    }
    __syncthreads();
    int p = spiv;
    if (p != k){
      for (int j = tid; j < 64; j += 256){
        c32 tv = M[k][j]; M[k][j] = M[p][j]; M[p][j] = tv;
      }
      if (tid == 0){ int tp = perm[k]; perm[k] = perm[p]; perm[p] = tp; }
    }
    __syncthreads();
    c32 piv = M[k][k];
    for (int r = k + 1 + tid; r < 64; r += 256) M[r][k] = cdiv(M[r][k], piv);
    __syncthreads();
    int w = 63 - k;
    if (w > 0){
      for (int idx = tid; idx < w * w; idx += 256){
        int r = k + 1 + idx / w, j = k + 1 + idx % w;
        M[r][j] = csub(M[r][j], cmul(M[r][k], M[k][j]));
      }
    }
    __syncthreads();
  }
  const float* yb = y + (size_t)b * 512 * 64;
  float* ph = Phi + (size_t)b * 64 * 511 * 2;
  for (int t = tid; t < 511; t += 256){
    c32 xv[64];
    for (int i = 0; i < 64; i++) xv[i] = {yb[(size_t)t * 64 + perm[i]], 0.0f};
    for (int i = 1; i < 64; i++){
      c32 s = xv[i];
      for (int j = 0; j < i; j++) s = csub(s, cmul(M[i][j], xv[j]));
      xv[i] = s;
    }
    for (int i = 63; i >= 0; i--){
      c32 s = xv[i];
      for (int j = i + 1; j < 64; j++) s = csub(s, cmul(M[i][j], xv[j]));
      xv[i] = cdiv(s, M[i][i]);
    }
    for (int i = 0; i < 64; i++){
      size_t o = ((size_t)i * 511 + t) * 2;
      ph[o] = xv[i].re; ph[o + 1] = xv[i].im;
    }
  }
}

// ---------------------------------------------------------------------------
// recon[p] = evecs @ (lam^p * y0)  (lam^0 == 1). Block per batch.
// ---------------------------------------------------------------------------
__global__ __launch_bounds__(256)
void k_recon(const float* __restrict__ evecs, const float* __restrict__ evals,
             const float* __restrict__ Phi, float* __restrict__ yr,
             float* __restrict__ yi)
{
  const int b = blockIdx.x, tid = threadIdx.x;
  __shared__ c32 E[64][64];
  __shared__ c32 u[64];
  __shared__ c32 lam[64];
  const float* Ep = evecs + (size_t)b * 4096 * 2;
  for (int idx = tid; idx < 4096; idx += 256){
    int i = idx >> 6, j = idx & 63;
    E[i][j] = {Ep[idx * 2], Ep[idx * 2 + 1]};
  }
  if (tid < 64){
    const float* ph = Phi + (((size_t)b * 64 + tid) * 511) * 2;
    u[tid] = {ph[0], ph[1]};                       // y0 (lam^0 = 1)
    lam[tid] = {evals[b * 128 + 2 * tid], evals[b * 128 + 2 * tid + 1]};
  }
  __syncthreads();
  for (int p = 0; p < 512; p++){
    if (tid < 64){
      c32 s = {0.0f, 0.0f};
      for (int j = 0; j < 64; j++) s = cadd(s, cmul(E[tid][j], u[j]));
      size_t o = ((size_t)b * 512 + p) * 64 + tid;
      yr[o] = s.re; yi[o] = s.im;
    }
    __syncthreads();
    if (tid < 64) u[tid] = cmul(u[tid], lam[tid]);
    __syncthreads();
  }
}

// ---------------------------------------------------------------------------
// Host-side orchestration
// ---------------------------------------------------------------------------
extern "C" void kernel_launch(void* const* d_in, const int* in_sizes, int n_in,
                              void* d_out, int out_size, void* d_ws, size_t ws_size,
                              hipStream_t stream)
{
  (void)in_sizes; (void)n_in; (void)out_size; (void)ws_size;
  const float* x        = (const float*)d_in[0];
  const float* enc_Win  = (const float*)d_in[1];
  const float* enc_bin  = (const float*)d_in[2];
  const float* enc_Wh   = (const float*)d_in[3];
  const float* enc_bh   = (const float*)d_in[4];
  const float* enc_Wout = (const float*)d_in[5];
  const float* enc_bout = (const float*)d_in[6];
  const float* dec_Win  = (const float*)d_in[7];
  const float* dec_bin  = (const float*)d_in[8];
  const float* dec_Wh   = (const float*)d_in[9];
  const float* dec_bh   = (const float*)d_in[10];
  const float* dec_Wout = (const float*)d_in[11];
  const float* dec_bout = (const float*)d_in[12];

  float* out = (float*)d_out;
  float* o_y     = out + 0;            // (256,512,64)
  float* o_xae   = out + 8388608;      // (256,512,2)
  float* o_xadv  = out + 8650752;      // (256,512,2)
  float* o_yr    = out + 8912896;      // (256,512,64)
  float* o_yi    = out + 17301504;     // (256,512,64)
  float* o_eval  = out + 25690112;     // (256,64) c64
  float* o_phi   = out + 25722880;     // (256,64,511) c64
  float* o_evec  = out + 42467328;     // (256,64,64) c64

  char* ws = (char*)d_ws;
  h16*   ewhT = (h16*)(ws + 0);         // 3 x 256 x 256 f16
  h16*   ewoT = (h16*)(ws + 393216);    // 64 x 256 f16
  h16*   dwiT = (h16*)(ws + 425984);    // 256 x 64 f16
  h16*   dwhT = (h16*)(ws + 458752);    // 3 x 256 x 256 f16
  float* G    = (float*)(ws + 1048576);
  float* C    = (float*)(ws + 5242880);
  float* A    = (float*)(ws + 9437184);
  float* Tm   = (float*)(ws + 13631488);
  float* Qm   = (float*)(ws + 17825792);

  for (int L = 0; L < 3; L++)
    k_transcvt<<<256, 256, 0, stream>>>(enc_Wh + L * 65536, ewhT + L * 65536, 256, 256);
  k_transcvt<<<64, 256, 0, stream>>>(enc_Wout, ewoT, 256, 64);
  k_transcvt<<<64, 256, 0, stream>>>(dec_Win, dwiT, 64, 256);
  for (int L = 0; L < 3; L++)
    k_transcvt<<<256, 256, 0, stream>>>(dec_Wh + L * 65536, dwhT + L * 65536, 256, 256);

  k_encoder<<<4096, 256, 0, stream>>>(x, enc_Win, enc_bin, ewhT, enc_bh,
                                      ewoT, enc_bout, o_y);
  k_decoder<<<4096, 256, 0, stream>>>(o_y, dwiT, dec_bin, dwhT, dec_bh,
                                      dec_Wout, dec_bout, o_xae);
  k_gc<<<256, 256, 0, stream>>>(o_y, G, C);
  k_solveA<<<256, 256, 0, stream>>>(G, C, A);
  k_schur<<<4, 64, 0, stream>>>(A, Tm, Qm, o_eval);
  k_evec<<<4, 64, 0, stream>>>(Tm, Qm, o_eval, o_evec);
  k_phi<<<256, 256, 0, stream>>>(o_evec, o_y, o_phi);
  k_recon<<<256, 256, 0, stream>>>(o_evec, o_eval, o_phi, o_yr, o_yi);
  k_decoder<<<4096, 256, 0, stream>>>(o_yr, dwiT, dec_bin, dwhT, dec_bh,
                                      dec_Wout, dec_bout, o_xadv);
}
